// SCADecoder_56642028700304
// MI455X (gfx1250) — matrix operand order repaired and checked by reading, not compiled, not run on hardware
//
#include <hip/hip_runtime.h>
#include <hip/hip_bf16.h>

// ---------------- problem constants ----------------
#define K_TOPICS 512
#define V_WORDS  50257
#define D_DIM    768
#define R_DIM    192
#define DHAT     960            // D + R (augmented dim), 960/32 = 30 k-steps
#define VP       50272          // V padded to multiple of 32
#define N_SINK   10
#define LAMBDA_MAHAL   0.3f
#define GAMMA_SINKHORN 0.3f

typedef __bf16 bf16_t;
typedef bf16_t bf16x16 __attribute__((ext_vector_type(16)));
typedef float  f32x8   __attribute__((ext_vector_type(8)));
typedef unsigned int uint32x4 __attribute__((ext_vector_type(4)));
typedef int          int32x8  __attribute__((ext_vector_type(8)));
typedef int          int32x4  __attribute__((ext_vector_type(4)));

// ---------------- small helpers ----------------
__device__ __forceinline__ bf16_t f2bf(float f) {
  unsigned u = __builtin_bit_cast(unsigned, f);
  unsigned r = u + 0x7FFFu + ((u >> 16) & 1u);       // round-to-nearest-even
  unsigned short h = (unsigned short)(r >> 16);
  return __builtin_bit_cast(bf16_t, h);
}
__device__ __forceinline__ float bf2f(bf16_t b) {
  unsigned short h = __builtin_bit_cast(unsigned short, b);
  unsigned u = ((unsigned)h) << 16;
  return __builtin_bit_cast(float, u);
}

// 4 contiguous elements: global (f32 or bf16) -> LDS bf16
__device__ __forceinline__ void g2lds4(const float* __restrict__ g, bf16_t* l) {
  const float4 v = *reinterpret_cast<const float4*>(g);
  l[0] = f2bf(v.x); l[1] = f2bf(v.y); l[2] = f2bf(v.z); l[3] = f2bf(v.w);
}
__device__ __forceinline__ void g2lds4(const bf16_t* __restrict__ g, bf16_t* l) {
  *reinterpret_cast<uint2*>(l) = *reinterpret_cast<const uint2*>(g);
}

// online logsumexp pair reduce over a 256-thread block; result in rm[0], rs[0]
__device__ __forceinline__ void block_lse_reduce(float* rm, float* rs, int tid,
                                                 float m, float s) {
  rm[tid] = m; rs[tid] = s; __syncthreads();
  for (int o = 128; o > 0; o >>= 1) {
    if (tid < o) {
      float m2 = rm[tid + o], s2 = rs[tid + o];
      float M = fmaxf(rm[tid], m2);
      float S = (M == -__builtin_inff()) ? 0.f
              : rs[tid] * __expf(rm[tid] - M) + s2 * __expf(m2 - M);
      rm[tid] = M; rs[tid] = S;
    }
    __syncthreads();
  }
}

// ---------------------------------------------------------------
// TDM: issue a 2-D tile DMA  global -> LDS (bf16 elements).
//   rows_tile rows of 32 contiguous elements, row stride
//   `stride_elems`, OOB rows (y >= rows_tensor) zero-filled by TDM.
// Descriptor packing per CDNA5 ISA.
// This toolchain's builtin takes 6 args (groups 0..3 + extra + cpol).
// ---------------------------------------------------------------
__device__ __forceinline__ void tdm_load_2d_bf16(const void* gptr, unsigned lds_off,
                                                 unsigned rows_tile,
                                                 unsigned rows_tensor,
                                                 unsigned stride_elems) {
  const unsigned long long ga = (unsigned long long)(size_t)gptr;
  uint32x4 g0;
  g0[0] = 1u;                                   // count=1, user descriptor
  g0[1] = lds_off;                              // LDS byte address
  g0[2] = (unsigned)ga;                         // global addr [31:0]
  g0[3] = (unsigned)(ga >> 32) | (2u << 30);    // addr [56:32] | type=2 (image)
  int32x8 g1;
  g1[0] = (int)(1u << 16);                      // data_size = 1 (2 bytes)
  g1[1] = (int)(32u << 16);                     // tensor_dim0[15:0]=32 in [31:16]
  g1[2] = (int)((rows_tensor & 0xFFFFu) << 16); // dim0 hi=0 | tensor_dim1 lo16
  g1[3] = (int)(((rows_tensor >> 16) & 0xFFFFu) // tensor_dim1 hi16
                | (32u << 16));                 // tile_dim0 = 32
  g1[4] = (int)(rows_tile & 0xFFFFu);           // tile_dim1 | tile_dim2=0
  g1[5] = (int)stride_elems;                    // tensor_dim0_stride lo32
  g1[6] = 0;                                    // stride hi | dim1_stride lo
  g1[7] = 0;
  const int32x4 z4 = {0, 0, 0, 0};              // groups 2/3 unused (2-D tile)
  const int32x8 z8 = {0, 0, 0, 0, 0, 0, 0, 0};
  __builtin_amdgcn_tensor_load_to_lds(g0, g1, z4, z4, z8, 0);
}

// ============================================================
// Generic bf16 WMMA GEMM:  C[M,N] = A[M,Kd] * BT[N,Kd]^T
// A row-major (lda), BT stored N-major/K-contiguous (ldb).
// Block = 256 threads = 8 waves -> 32x64 C tile (2x4 waves of 16x16).
// ============================================================
template <typename TA, typename TB, bool OUT_BF16>
__global__ void __launch_bounds__(256)
k_gemm_bt(const TA* __restrict__ A, int lda,
          const TB* __restrict__ BT, int ldb,
          void* __restrict__ Cout, int ldc,
          int M, int N, int Kd)
{
  __shared__ bf16_t lsA[32 * 32];
  __shared__ bf16_t lsB[64 * 32];
  const int tid  = threadIdx.x;
  const int lane = tid & 31;
  const int wave = tid >> 5;
  const int tm = wave >> 2;            // 0..1  (M sub-tile)
  const int tn = wave & 3;             // 0..3  (N sub-tile)
  const int m0 = blockIdx.x * 32;
  const int n0 = blockIdx.y * 64;

  const int ar = tid >> 3, ac = (tid & 7) * 4;      // A: 32x32, 4 elems/thread
  const int br = tid >> 2, bc = (tid & 3) * 8;      // B: 64x32, 8 elems/thread
  const int arow = min(m0 + ar, M - 1);
  const int brow = min(n0 + br, N - 1);

  f32x8 acc = {0.f, 0.f, 0.f, 0.f, 0.f, 0.f, 0.f, 0.f};

  for (int k0 = 0; k0 < Kd; k0 += 32) {
    __syncthreads();
    g2lds4(A  + (size_t)arow * lda + (k0 + ac),     &lsA[ar * 32 + ac]);
    g2lds4(BT + (size_t)brow * ldb + (k0 + bc),     &lsB[br * 32 + bc]);
    g2lds4(BT + (size_t)brow * ldb + (k0 + bc + 4), &lsB[br * 32 + bc + 4]);
    if (k0 + 32 < Kd) {   // prefetch next K slab -> global_prefetch_b8
      __builtin_prefetch(A  + (size_t)arow * lda + (k0 + 32 + ac), 0, 1);
      __builtin_prefetch(BT + (size_t)brow * ldb + (k0 + 32 + bc), 0, 1);
    }
    __syncthreads();

    bf16x16 fa, fb;
    const int ml = lane & 15;
    const int kb = (lane < 16) ? 0 : 8;
    #pragma unroll
    for (int j = 0; j < 8; ++j) {
      int kk = (j < 4) ? (kb + 2 * j) : (16 + kb + 2 * (j - 4));
      fa[2 * j]     = lsA[(tm * 16 + ml) * 32 + kk];
      fa[2 * j + 1] = lsA[(tm * 16 + ml) * 32 + kk + 1];
    }
    const int kh = (lane < 16) ? 0 : 16;
    #pragma unroll
    for (int j = 0; j < 8; ++j) {
      int kk = (j < 4) ? (kb + 2 * j) : (16 + kb + 2 * (j - 4));
      fb[2 * j]     = lsB[(tn * 16 + ml) * 32 + kk];
      fb[2 * j + 1] = lsB[(tn * 16 + ml) * 32 + kk + 1];
    }
    acc = __builtin_amdgcn_wmma_f32_16x16x32_bf16(false, fa, false, fb,
                                                  (short)0, acc, false, false);
  }

  const int crow0 = m0 + tm * 16 + ((lane < 16) ? 0 : 8);
  const int ccol  = n0 + tn * 16 + (lane & 15);
  if (ccol < N) {
    #pragma unroll
    for (int i = 0; i < 8; ++i) {
      int r = crow0 + i;
      if (r < M) {
        if (OUT_BF16) ((bf16_t*)Cout)[(size_t)r * ldc + ccol] = f2bf(acc[i]);
        else          ((float*) Cout)[(size_t)r * ldc + ccol] = acc[i];
      }
    }
  }
}

// ============================================================
// Stage 3: fused similarity GEMM + Mahalanobis/direct epilogue.
//   out_full = Ahat[512,960] . What[v,960]^T  (AB term)
//   accumulator snapshot at k=768 gives a.w (direct cosine term).
// TDM-fed, double-buffered LDS; 8 waves (2Mx4N), each wave owns a
// 32x32 register tile -> 4 WMMAs / wave / k-step with frag reuse.
// Block tile 64(M) x 128(N); only wave 0 touches memory for tiles.
// ============================================================
__global__ void __launch_bounds__(256)
k_stage3(const bf16_t* __restrict__ Ahat, const bf16_t* __restrict__ What,
         const float* __restrict__ AA, const float* __restrict__ BB,
         const float* __restrict__ bias, const float* __restrict__ C,
         float* __restrict__ DIRECT, float* __restrict__ MAHAL)
{
  __shared__ bf16_t lsA[2][64 * 32];     // 2 x 4KB
  __shared__ bf16_t lsB[2][128 * 32];    // 2 x 8KB
  const int tid  = threadIdx.x;
  const int lane = tid & 31;
  const int wave = tid >> 5;
  const int wm = wave >> 2;              // 0..1 : 32-row slice of M tile
  const int wn = wave & 3;               // 0..3 : 32-col slice of N tile
  const int m0 = blockIdx.y * 64;        // topic tile (M = 512 exact)
  const int n0 = blockIdx.x * 128;       // word tile

  const unsigned rowsB_tensor = (unsigned)(V_WORDS - n0);  // TDM zero-fills tail

  // prologue: DMA first K-slab into buffer 0 (wave 0 only; EXEC-independent op)
  if (wave == 0) {
    tdm_load_2d_bf16(Ahat + (size_t)m0 * DHAT, (unsigned)(size_t)&lsA[0][0],
                     64u, 64u, DHAT);
    tdm_load_2d_bf16(What + (size_t)n0 * DHAT, (unsigned)(size_t)&lsB[0][0],
                     128u, rowsB_tensor, DHAT);
  }

  f32x8 acc[2][2]    = {};
  f32x8 accDir[2][2] = {};

  const int ml = lane & 15;
  const int kb = (lane < 16) ? 0 : 8;
  const int kh = (lane < 16) ? 0 : 16;

  int cur = 0;
  const int KSTEPS = DHAT / 32;          // 30
  for (int ks = 0; ks < KSTEPS; ++ks) {
    if (wave == 0) {
      if (ks + 1 < KSTEPS) {
        const int kn = (ks + 1) * 32;
        tdm_load_2d_bf16(Ahat + (size_t)m0 * DHAT + kn,
                         (unsigned)(size_t)&lsA[cur ^ 1][0], 64u, 64u, DHAT);
        tdm_load_2d_bf16(What + (size_t)n0 * DHAT + kn,
                         (unsigned)(size_t)&lsB[cur ^ 1][0], 128u, rowsB_tensor, DHAT);
        // TDM completes in-order per wave: <=2 outstanding means the two
        // loads for `cur` have landed, only the just-issued pair pending.
        __builtin_amdgcn_s_wait_tensorcnt(2);
      } else {
        __builtin_amdgcn_s_wait_tensorcnt(0);
      }
    }
    __syncthreads();                     // cur buffers resident for all waves

    bf16x16 fa[2], fb[2];
    #pragma unroll
    for (int t = 0; t < 2; ++t) {
      const int row = (wm * 32 + t * 16 + ml) * 32;
      #pragma unroll
      for (int j = 0; j < 8; ++j) {
        int kk = (j < 4) ? (kb + 2 * j) : (16 + kb + 2 * (j - 4));
        fa[t][2 * j]     = lsA[cur][row + kk];
        fa[t][2 * j + 1] = lsA[cur][row + kk + 1];
      }
    }
    #pragma unroll
    for (int t = 0; t < 2; ++t) {
      const int row = (wn * 32 + t * 16 + ml) * 32;
      #pragma unroll
      for (int j = 0; j < 8; ++j) {
        int kk = (j < 4) ? (kb + 2 * j) : (16 + kb + 2 * (j - 4));
        fb[t][2 * j]     = lsB[cur][row + kk];
        fb[t][2 * j + 1] = lsB[cur][row + kk + 1];
      }
    }
    #pragma unroll
    for (int i = 0; i < 2; ++i)
      #pragma unroll
      for (int j = 0; j < 2; ++j)
        acc[i][j] = __builtin_amdgcn_wmma_f32_16x16x32_bf16(
            false, fa[i], false, fb[j], (short)0, acc[i][j], false, false);

    if ((ks + 1) * 32 == D_DIM) {        // snapshot: plain a.w after K=768
      #pragma unroll
      for (int i = 0; i < 2; ++i)
        #pragma unroll
        for (int j = 0; j < 2; ++j)
          accDir[i][j] = acc[i][j];
    }
    __syncthreads();                     // reads done before buffer reuse
    cur ^= 1;
  }

  const float scale = C[0];
  #pragma unroll
  for (int i = 0; i < 2; ++i) {
    const int crow0 = m0 + wm * 32 + i * 16 + ((lane < 16) ? 0 : 8);
    #pragma unroll
    for (int j = 0; j < 2; ++j) {
      const int ccol = n0 + wn * 32 + j * 16 + (lane & 15);
      if (ccol < V_WORDS) {
        const float bb = BB[ccol];
        #pragma unroll
        for (int e = 0; e < 8; ++e) {
          int r = crow0 + e;
          float mah = fmaxf(AA[r] + bb - 2.f * acc[i][j][e], 0.f);
          DIRECT[(size_t)r * VP + ccol] = scale * accDir[i][j][e] + bias[r];
          MAHAL [(size_t)r * VP + ccol] = mah;
        }
      }
    }
  }
}

// ============================================================
// Prep kernels
// ============================================================
__global__ void k_consts(const float* le, const float* ls, float* C) {
  if (threadIdx.x == 0 && blockIdx.x == 0) {
    float scale = fminf(expf(ls[0]), 20.f);
    float x = le[0];
    float sp  = (x > 20.f) ? x : log1pf(expf(x));   // softplus
    float eps = sp + 0.001f;
    C[0] = scale; C[1] = eps; C[2] = 1.f / eps;
  }
}

__global__ void k_transpose_u(const float* __restrict__ U, float* __restrict__ Ut) {
  int idx = blockIdx.x * 256 + threadIdx.x;
  if (idx < D_DIM * R_DIM) {
    int d = idx / R_DIM, r = idx % R_DIM;
    Ut[(size_t)r * D_DIM + d] = U[idx];
  }
}

// anchors: normalize, build Ahat=[a ; U^T a] (bf16), AA = 1 + |U^T a|^2
__global__ void __launch_bounds__(256)
k_prep_anchors(const float* __restrict__ anchors, const float* __restrict__ U,
               bf16_t* __restrict__ Ahat, float* __restrict__ AA)
{
  __shared__ float sa[D_DIM];
  __shared__ float red[256];
  const int k = blockIdx.x, tid = threadIdx.x;
  float ss = 0.f;
  for (int d = tid; d < D_DIM; d += 256) {
    float x = anchors[(size_t)k * D_DIM + d]; sa[d] = x; ss += x * x;
  }
  red[tid] = ss; __syncthreads();
  for (int o = 128; o > 0; o >>= 1) { if (tid < o) red[tid] += red[tid + o]; __syncthreads(); }
  const float rn = 1.f / fmaxf(sqrtf(red[0]), 1e-12f);
  __syncthreads();
  for (int d = tid; d < D_DIM; d += 256) {
    float x = sa[d] * rn; sa[d] = x;
    Ahat[(size_t)k * DHAT + d] = f2bf(x);
  }
  __syncthreads();
  float s = 0.f;
  if (tid < R_DIM) {
    for (int d = 0; d < D_DIM; ++d) s += sa[d] * U[(size_t)d * R_DIM + tid];
    Ahat[(size_t)k * DHAT + D_DIM + tid] = f2bf(s);
  }
  red[tid] = (tid < R_DIM) ? s * s : 0.f; __syncthreads();
  for (int o = 128; o > 0; o >>= 1) { if (tid < o) red[tid] += red[tid + o]; __syncthreads(); }
  if (tid == 0) AA[k] = 1.f + red[0];
}

// normalize projected words (P bf16 [V,768]) -> What[:, 0:768]
__global__ void __launch_bounds__(256)
k_norm_words(const bf16_t* __restrict__ P, bf16_t* __restrict__ What)
{
  __shared__ float red[256];
  const int v = blockIdx.x, tid = threadIdx.x;
  float xs[3]; float ss = 0.f;
  #pragma unroll
  for (int i = 0; i < 3; ++i) {
    int d = tid + i * 256;
    float x = bf2f(P[(size_t)v * D_DIM + d]); xs[i] = x; ss += x * x;
  }
  red[tid] = ss; __syncthreads();
  for (int o = 128; o > 0; o >>= 1) { if (tid < o) red[tid] += red[tid + o]; __syncthreads(); }
  float rn = 1.f / fmaxf(sqrtf(red[0]), 1e-12f);
  #pragma unroll
  for (int i = 0; i < 3; ++i) {
    int d = tid + i * 256;
    What[(size_t)v * DHAT + d] = f2bf(xs[i] * rn);
  }
}

// S[V,192] -> What[:,768:960] bf16, BB[v] = 1 + |S_v|^2
__global__ void __launch_bounds__(256)
k_bb(const float* __restrict__ S, bf16_t* __restrict__ What, float* __restrict__ BB)
{
  __shared__ float red[256];
  const int v = blockIdx.x, tid = threadIdx.x;
  float s2 = 0.f;
  if (tid < R_DIM) {
    float x = S[(size_t)v * R_DIM + tid];
    What[(size_t)v * DHAT + D_DIM + tid] = f2bf(x);
    s2 = x * x;
  }
  red[tid] = s2; __syncthreads();
  for (int o = 128; o > 0; o >>= 1) { if (tid < o) red[tid] += red[tid + o]; __syncthreads(); }
  if (tid == 0) BB[v] = 1.f + red[0];
}

__global__ void k_init_logv(float* LOGV) {
  int i = blockIdx.x * 256 + threadIdx.x;
  if (i < VP) LOGV[i] = 0.f;
}

// ============================================================
// Sinkhorn passes over L2-resident MAHAL (logK = -mahal/eps)
// ============================================================
__global__ void __launch_bounds__(256)
k_sink_row(const float* __restrict__ MAHAL, const float* __restrict__ LOGV,
           const float* __restrict__ C, float* __restrict__ LOGU)
{
  __shared__ float rm[256], rs[256];
  const int k = blockIdx.x, tid = threadIdx.x;
  const float ie = C[2];
  float m = -__builtin_inff(), s = 0.f;
  for (int v = tid; v < V_WORDS; v += 256) {
    float x = -MAHAL[(size_t)k * VP + v] * ie + LOGV[v];
    if (x > m) { s = s * __expf(m - x) + 1.f; m = x; } else s += __expf(x - m);
  }
  block_lse_reduce(rm, rs, tid, m, s);
  if (tid == 0) LOGU[k] = -(rm[0] + __logf(rs[0]));
}

__global__ void __launch_bounds__(256)
k_sink_col(const float* __restrict__ MAHAL, const float* __restrict__ LOGU,
           const float* __restrict__ C, float* __restrict__ LOGV)
{
  __shared__ float slu[K_TOPICS];
  const int tid = threadIdx.x;
  for (int i = tid; i < K_TOPICS; i += 256) slu[i] = LOGU[i];
  __syncthreads();
  const int v = blockIdx.x * 256 + tid;
  if (v >= V_WORDS) return;
  const float ie = C[2];
  float m = -__builtin_inff(), s = 0.f;
  for (int k = 0; k < K_TOPICS; ++k) {
    float x = -MAHAL[(size_t)k * VP + v] * ie + slu[k];
    if (x > m) { s = s * __expf(m - x) + 1.f; m = x; } else s += __expf(x - m);
  }
  LOGV[v] = -(m + __logf(s));
}

// ============================================================
// Finalize: logits = direct - L*mahal + G*(-mahal/eps + lu + lv),
//           clip +-30, row softmax -> d_out [512, 50257]
// ============================================================
__global__ void __launch_bounds__(256)
k_finalize(const float* __restrict__ DIRECT, const float* __restrict__ MAHAL,
           const float* __restrict__ LOGU, const float* __restrict__ LOGV,
           const float* __restrict__ C, float* __restrict__ out)
{
  __shared__ float rm[256], rs[256];
  const int k = blockIdx.x, tid = threadIdx.x;
  const float ie = C[2];
  const float lu = LOGU[k];
  float m = -__builtin_inff(), s = 0.f;
  for (int v = tid; v < V_WORDS; v += 256) {
    float mah = MAHAL[(size_t)k * VP + v];
    float lg = DIRECT[(size_t)k * VP + v] - LAMBDA_MAHAL * mah
             + GAMMA_SINKHORN * (-mah * ie + lu + LOGV[v]);
    lg = fminf(fmaxf(lg, -30.f), 30.f);
    if (lg > m) { s = s * __expf(m - lg) + 1.f; m = lg; } else s += __expf(lg - m);
  }
  block_lse_reduce(rm, rs, tid, m, s);
  const float M = rm[0];
  const float inv = 1.f / rs[0];
  for (int v = tid; v < V_WORDS; v += 256) {
    float mah = MAHAL[(size_t)k * VP + v];
    float lg = DIRECT[(size_t)k * VP + v] - LAMBDA_MAHAL * mah
             + GAMMA_SINKHORN * (-mah * ie + lu + LOGV[v]);
    lg = fminf(fmaxf(lg, -30.f), 30.f);
    out[(size_t)k * V_WORDS + v] = __expf(lg - M) * inv;
  }
}

// ============================================================
// Host launcher
// ============================================================
extern "C" void kernel_launch(void* const* d_in, const int* in_sizes, int n_in,
                              void* d_out, int out_size, void* d_ws, size_t ws_size,
                              hipStream_t stream)
{
  const float* word_embeds = (const float*)d_in[0];   // [V, D]
  const float* anchors     = (const float*)d_in[1];   // [K, D]
  const float* M_U         = (const float*)d_in[2];   // [D, R]
  const float* log_eps     = (const float*)d_in[3];   // [1]
  const float* log_scale   = (const float*)d_in[4];   // [1]
  const float* topic_bias  = (const float*)d_in[5];   // [K]
  const float* word_proj_w = (const float*)d_in[6];   // [D, D]

  char* wsb = (char*)d_ws;
  size_t off = 0;
  auto alloc = [&](size_t bytes) -> void* {
    void* p = (void*)(wsb + off);
    off += (bytes + 255) & ~(size_t)255;
    return p;
  };

  bf16_t* Ahat   = (bf16_t*)alloc((size_t)K_TOPICS * DHAT * sizeof(bf16_t));
  bf16_t* What   = (bf16_t*)alloc((size_t)VP * DHAT * sizeof(bf16_t));
  float*  AA     = (float*) alloc((size_t)K_TOPICS * sizeof(float));
  float*  BB     = (float*) alloc((size_t)VP * sizeof(float));
  float*  LOGU   = (float*) alloc((size_t)K_TOPICS * sizeof(float));
  float*  LOGV   = (float*) alloc((size_t)VP * sizeof(float));
  float*  CONSTS = (float*) alloc(16 * sizeof(float));
  float*  Ut     = (float*) alloc((size_t)R_DIM * D_DIM * sizeof(float));
  float*  DIRECT = (float*) alloc((size_t)K_TOPICS * VP * sizeof(float));
  float*  MAHAL  = (float*) alloc((size_t)K_TOPICS * VP * sizeof(float));

  // dead-intermediate aliasing: P (bf16 [V,768], 77MB) lives in DIRECT (103MB)
  // until stage 3; S (f32 [V,192], 39MB) lives in MAHAL until k_bb.
  bf16_t* P = (bf16_t*)DIRECT;
  float*  S = MAHAL;

  const int MT_V = (V_WORDS + 31) / 32;     // 1571 row tiles over words

  k_consts<<<1, 1, 0, stream>>>(log_eps, log_scale, CONSTS);
  k_transpose_u<<<(D_DIM * R_DIM + 255) / 256, 256, 0, stream>>>(M_U, Ut);
  k_prep_anchors<<<K_TOPICS, 256, 0, stream>>>(anchors, M_U, Ahat, AA);

  // P = E @ W^T  (W row-major already is BT layout: BT[d][j] = W[d][j])
  {
    dim3 g(MT_V, D_DIM / 64);
    k_gemm_bt<float, float, true><<<g, 256, 0, stream>>>(
        word_embeds, D_DIM, word_proj_w, D_DIM, (void*)P, D_DIM,
        V_WORDS, D_DIM, D_DIM);
  }
  k_norm_words<<<V_WORDS, 256, 0, stream>>>(P, What);

  // S = w @ U   (A = What[:, :768] bf16, BT = U^T [192,768] f32)
  {
    dim3 g(MT_V, R_DIM / 64);
    k_gemm_bt<bf16_t, float, false><<<g, 256, 0, stream>>>(
        What, DHAT, Ut, D_DIM, (void*)S, R_DIM,
        V_WORDS, R_DIM, D_DIM);
  }
  k_bb<<<V_WORDS, 256, 0, stream>>>(S, What, BB);

  // fused similarity GEMM + epilogue (TDM-fed, overwrites aliased P/S)
  {
    dim3 g((V_WORDS + 127) / 128, K_TOPICS / 64);
    k_stage3<<<g, 256, 0, stream>>>(Ahat, What, AA, BB, topic_bias, CONSTS,
                                    DIRECT, MAHAL);
  }

  // Sinkhorn (L2-resident MAHAL, 103MB < 192MB L2)
  k_init_logv<<<(VP + 255) / 256, 256, 0, stream>>>(LOGV);
  for (int it = 0; it < N_SINK; ++it) {
    k_sink_row<<<K_TOPICS, 256, 0, stream>>>(MAHAL, LOGV, CONSTS, LOGU);
    k_sink_col<<<(V_WORDS + 255) / 256, 256, 0, stream>>>(MAHAL, LOGU, CONSTS, LOGV);
  }

  k_finalize<<<K_TOPICS, 256, 0, stream>>>(DIRECT, MAHAL, LOGU, LOGV, CONSTS,
                                           (float*)d_out);
  (void)in_sizes; (void)n_in; (void)out_size; (void)ws_size;
}
